// MultiHeadAttention_66580583022857
// MI455X (gfx1250) — matrix operand order repaired and checked by reading, not compiled, not run on hardware
//
#include <hip/hip_runtime.h>
#include <hip/hip_bf16.h>
#include <stdint.h>

#define HEADS 16
#define EMBED 1024
#define BATCH 2
#define SEQ   2048
#define HDIM  64

typedef __bf16 bf16;
typedef bf16  v16bf  __attribute__((ext_vector_type(16)));
typedef float v8f    __attribute__((ext_vector_type(8)));
typedef unsigned int uint4v __attribute__((ext_vector_type(4)));

union Frag {
  v16bf  bf;
  uint4v q[2];
};

__device__ __forceinline__ v8f wmma_bf16(const Frag& a, const Frag& b, v8f c) {
  return __builtin_amdgcn_wmma_f32_16x16x32_bf16(false, a.bf, false, b.bf,
                                                 (short)0, c, false, false);
}

// ---------------------------------------------------------------------------
// fp32 -> bf16 conversion
// ---------------------------------------------------------------------------
__global__ __launch_bounds__(256)
void cvt_f32_bf16(const float* __restrict__ src, unsigned short* __restrict__ dst, int n) {
  int i = blockIdx.x * 256 + threadIdx.x;
  if (i < n) {
    union { bf16 h; unsigned short u; } cv;
    cv.h = (bf16)src[i];
    dst[i] = cv.u;
  }
}

// ---------------------------------------------------------------------------
// Shared 64x64-per-wave GEMM core:  C(64x64) += A[M,K] x W[N,K]^T tile.
// 16 WMMAs per 16 b128 loads per K=32 step (1:1 VMEM:WMMA issue ratio).
// ---------------------------------------------------------------------------
__device__ __forceinline__ void gemm_core_64x64(const bf16* __restrict__ A,
                                                const bf16* __restrict__ W,
                                                int mBase, int nBase,
                                                int lm, int g, int K,
                                                v8f (&acc)[4][4]) {
  for (int k0 = 0; k0 < K; k0 += 32) {
    Frag a[4], b[4];
#pragma unroll
    for (int i = 0; i < 4; ++i) {
      const bf16* p = A + (size_t)(mBase + i * 16 + lm) * K + k0 + g * 8;
      a[i].q[0] = *(const uint4v*)p;
      a[i].q[1] = *(const uint4v*)(p + 16);
    }
#pragma unroll
    for (int j = 0; j < 4; ++j) {
      const bf16* p = W + (size_t)(nBase + j * 16 + lm) * K + k0 + g * 8;
      b[j].q[0] = *(const uint4v*)p;
      b[j].q[1] = *(const uint4v*)(p + 16);
    }
#pragma unroll
    for (int i = 0; i < 4; ++i)
#pragma unroll
      for (int j = 0; j < 4; ++j)
        acc[i][j] = wmma_bf16(a[i], b[j], acc[i][j]);
  }
}

// ---------------------------------------------------------------------------
// QKV GEMM: [4096,1024] x [3072,1024]^T, scatter into Q/K [b,h,t,d] and
// V transposed [b,h,d,t] (bf16).  Q is pre-scaled by 8 (reference multiplies
// scores by sqrt(head_dim)=8).  Block = 8 waves (2M x 4N) -> 128x256 tile.
// ---------------------------------------------------------------------------
__global__ __launch_bounds__(256)
void gemm_qkv_kernel(const unsigned short* __restrict__ Araw,
                     const unsigned short* __restrict__ Wraw,
                     unsigned short* __restrict__ qraw,
                     unsigned short* __restrict__ kraw,
                     unsigned short* __restrict__ vraw) {
  const bf16* A  = (const bf16*)Araw;
  const bf16* W  = (const bf16*)Wraw;
  bf16* qb = (bf16*)qraw;
  bf16* kb = (bf16*)kraw;
  bf16* vt = (bf16*)vraw;

  const int lane = threadIdx.x & 31;
  const int wv   = threadIdx.x >> 5;
  const int lm   = lane & 15;
  const int g    = lane >> 4;
  const int mBase = blockIdx.x * 128 + (wv & 1) * 64;
  const int nBase = blockIdx.y * 256 + (wv >> 1) * 64;

  v8f acc[4][4] = {};
  gemm_core_64x64(A, W, mBase, nBase, lm, g, EMBED, acc);

  // Epilogue: column n = dd*48 + kind*16 + h  (reference reshape (d,3,h)).
#pragma unroll
  for (int j = 0; j < 4; ++j) {
    const int n0   = nBase + j * 16;      // multiple of 16
    const int kind = (n0 >> 4) % 3;
    const int dd   = n0 / 48;
    const int h    = lm;
#pragma unroll
    for (int i = 0; i < 4; ++i) {
#pragma unroll
      for (int v = 0; v < 8; ++v) {
        const int row = mBase + i * 16 + g * 8 + v;
        const int bb  = row >> 11;          // / SEQ
        const int t   = row & 2047;         // % SEQ
        if (kind == 0)       // Q, pre-scaled by 8
          qb[(((size_t)(bb * HEADS + h) * SEQ + t) << 6) + dd] = (bf16)(acc[i][j][v] * 8.0f);
        else if (kind == 1)  // K
          kb[(((size_t)(bb * HEADS + h) * SEQ + t) << 6) + dd] = (bf16)acc[i][j][v];
        else                 // V transposed [b,h,d,t]
          vt[(((size_t)(bb * HEADS + h) * HDIM + dd) << 11) + t] = (bf16)acc[i][j][v];
      }
    }
  }
}

// ---------------------------------------------------------------------------
// Flash attention: one wave handles 32 query rows (2 row-tiles) of one (b,h).
// K/V fragments loaded once per 32-key step, reused by both row-tiles:
// 16 WMMAs per 16 b128 loads.  Scores already carry the *8 scale (in Q).
// ---------------------------------------------------------------------------
__global__ __launch_bounds__(256)
void attn_kernel(const unsigned short* __restrict__ qraw,
                 const unsigned short* __restrict__ kraw,
                 const unsigned short* __restrict__ vraw,
                 unsigned short* __restrict__ oraw) {
  __shared__ bf16 sm[8 * 2 * 16 * 32];   // per-wave: two 16x32 bf16 P tiles

  const bf16* qb = (const bf16*)qraw;
  const bf16* kb = (const bf16*)kraw;
  const bf16* vt = (const bf16*)vraw;
  bf16* ob = (bf16*)oraw;

  const int lane = threadIdx.x & 31;
  const int wv   = threadIdx.x >> 5;
  const int lm   = lane & 15;
  const int g    = lane >> 4;
  const int wid  = blockIdx.x * 8 + wv;
  const int bh   = wid >> 6;          // (b*16+h)
  const int t0   = (wid & 63) << 5;   // 32-row query tile base

  const bf16* qrow  = qb + ((size_t)bh * SEQ + t0) * HDIM;
  const bf16* kbase = kb + (size_t)bh * SEQ * HDIM;
  const bf16* vbase = vt + (size_t)bh * HDIM * SEQ;
  bf16* ptile = sm + wv * (2 * 16 * 32);

  // Q fragments: 2 row-tiles x 2 dd-halves, resident for the whole loop.
  Frag aq[2][2];
#pragma unroll
  for (int i = 0; i < 2; ++i)
#pragma unroll
    for (int hh = 0; hh < 2; ++hh) {
      const bf16* p = qrow + (i * 16 + lm) * HDIM + hh * 32 + g * 8;
      aq[i][hh].q[0] = *(const uint4v*)p;
      aq[i][hh].q[1] = *(const uint4v*)(p + 16);
    }

  float mi[2][8], li[2][8];
#pragma unroll
  for (int i = 0; i < 2; ++i)
#pragma unroll
    for (int v = 0; v < 8; ++v) { mi[i][v] = -1e30f; li[i][v] = 0.f; }
  v8f o[2][4] = {};

  for (int j0 = 0; j0 < SEQ; j0 += 32) {
    if (j0 + 32 < SEQ) {   // prefetch next K/V tiles (global_prefetch_b8)
      __builtin_prefetch(kbase + (size_t)(j0 + 32 + lm) * HDIM, 0, 0);
      __builtin_prefetch(vbase + (size_t)lm * SEQ + j0 + 32, 0, 0);
    }

    // ---- K fragments for 32 keys (2 j-subtiles x 2 dd-halves) ----
    Frag bk[2][2];
#pragma unroll
    for (int jt = 0; jt < 2; ++jt)
#pragma unroll
      for (int hh = 0; hh < 2; ++hh) {
        const bf16* p = kbase + (size_t)(j0 + jt * 16 + lm) * HDIM + hh * 32 + g * 8;
        bk[jt][hh].q[0] = *(const uint4v*)p;
        bk[jt][hh].q[1] = *(const uint4v*)(p + 16);
      }

    // ---- S = Q @ K^T : 8 WMMAs (K frags reused by both row-tiles) ----
    v8f s[2][2];
#pragma unroll
    for (int i = 0; i < 2; ++i)
#pragma unroll
      for (int jt = 0; jt < 2; ++jt) {
        v8f accS = {};
#pragma unroll
        for (int hh = 0; hh < 2; ++hh)
          accS = wmma_bf16(aq[i][hh], bk[jt][hh], accS);
        s[i][jt] = accS;
      }

    // ---- online softmax per row-tile ----
#pragma unroll
    for (int i = 0; i < 2; ++i) {
#pragma unroll
      for (int v = 0; v < 8; ++v) {
        float s0 = s[i][0][v];
        float s1 = s[i][1][v];
        float rmax = fmaxf(s0, s1);
#pragma unroll
        for (int msk = 1; msk < 16; msk <<= 1)
          rmax = fmaxf(rmax, __shfl_xor(rmax, msk, 32));
        float mnew  = fmaxf(mi[i][v], rmax);
        float alpha = __expf(mi[i][v] - mnew);
        mi[i][v] = mnew;
        float p0 = __expf(s0 - mnew);
        float p1 = __expf(s1 - mnew);
        float rs = p0 + p1;
#pragma unroll
        for (int msk = 1; msk < 16; msk <<= 1)
          rs += __shfl_xor(rs, msk, 32);
        li[i][v] = li[i][v] * alpha + rs;
#pragma unroll
        for (int nt = 0; nt < 4; ++nt) o[i][nt][v] *= alpha;
        const int m = g * 8 + v;
        ptile[i * 512 + m * 32 + lm]      = (bf16)p0;
        ptile[i * 512 + m * 32 + 16 + lm] = (bf16)p1;
      }
    }

    // LDS RAW: wait for P stores before fragment reload
    asm volatile("s_wait_dscnt 0" ::: "memory");

    // ---- reload P in A-fragment layout (lane row = lm) ----
    Frag pa[2];
#pragma unroll
    for (int i = 0; i < 2; ++i) {
      const bf16* p = ptile + i * 512 + lm * 32 + g * 8;
      pa[i].q[0] = *(const uint4v*)p;
      pa[i].q[1] = *(const uint4v*)(p + 16);
    }

    // ---- V fragments (loaded once, reused by both row-tiles) ----
    Frag bv[4];
#pragma unroll
    for (int nt = 0; nt < 4; ++nt) {
      const bf16* p = vbase + (size_t)(nt * 16 + lm) * SEQ + j0 + g * 8;
      bv[nt].q[0] = *(const uint4v*)p;
      bv[nt].q[1] = *(const uint4v*)(p + 16);
    }

    // ---- O += P @ V : 8 WMMAs ----
#pragma unroll
    for (int i = 0; i < 2; ++i)
#pragma unroll
      for (int nt = 0; nt < 4; ++nt)
        o[i][nt] = wmma_bf16(pa[i], bv[nt], o[i][nt]);
  }

  // ---- normalize and store to [b, t, h*64+dd] bf16 ----
  const int b = bh >> 4;
  const int h = bh & 15;
#pragma unroll
  for (int i = 0; i < 2; ++i) {
#pragma unroll
    for (int v = 0; v < 8; ++v) {
      float inv = 1.0f / li[i][v];
      const int t = t0 + i * 16 + g * 8 + v;
#pragma unroll
      for (int nt = 0; nt < 4; ++nt)
        ob[((size_t)(b * SEQ + t)) * EMBED + h * HDIM + nt * 16 + lm] =
            (bf16)(o[i][nt][v] * inv);
    }
  }
}

// ---------------------------------------------------------------------------
// Output projection: [4096,1024] x [1024,1024]^T -> fp32 out
// ---------------------------------------------------------------------------
__global__ __launch_bounds__(256)
void gemm_out_kernel(const unsigned short* __restrict__ Araw,
                     const unsigned short* __restrict__ Wraw,
                     float* __restrict__ out) {
  const bf16* A = (const bf16*)Araw;
  const bf16* W = (const bf16*)Wraw;

  const int lane = threadIdx.x & 31;
  const int wv   = threadIdx.x >> 5;
  const int lm   = lane & 15;
  const int g    = lane >> 4;
  const int mBase = blockIdx.x * 128 + (wv & 1) * 64;
  const int nBase = blockIdx.y * 256 + (wv >> 1) * 64;

  v8f acc[4][4] = {};
  gemm_core_64x64(A, W, mBase, nBase, lm, g, EMBED, acc);

#pragma unroll
  for (int j = 0; j < 4; ++j) {
    const int n0 = nBase + j * 16;
#pragma unroll
    for (int i = 0; i < 4; ++i) {
#pragma unroll
      for (int v = 0; v < 8; ++v) {
        const int row = mBase + i * 16 + g * 8 + v;
        out[(size_t)row * EMBED + n0 + lm] = acc[i][j][v];
      }
    }
  }
}

// ---------------------------------------------------------------------------
// Host launcher
// ---------------------------------------------------------------------------
extern "C" void kernel_launch(void* const* d_in, const int* in_sizes, int n_in,
                              void* d_out, int out_size, void* d_ws, size_t ws_size,
                              hipStream_t stream) {
  (void)in_sizes; (void)n_in; (void)out_size; (void)ws_size;

  const float* x      = (const float*)d_in[0];
  const float* w_qkv  = (const float*)d_in[1];
  const float* w_out  = (const float*)d_in[2];
  float*       out    = (float*)d_out;
  char*        ws     = (char*)d_ws;

  const size_t N_X    = (size_t)BATCH * SEQ * EMBED;          // 4194304
  const size_t N_WQKV = (size_t)3 * EMBED * EMBED;            // 3145728
  const size_t N_WOUT = (size_t)EMBED * EMBED;                // 1048576
  const size_t N_HBUF = (size_t)BATCH * HEADS * SEQ * HDIM;   // 4194304

  unsigned short* xb   = (unsigned short*)(ws);
  unsigned short* wqb  = (unsigned short*)(ws + 2 * N_X);
  unsigned short* wob  = (unsigned short*)(ws + 2 * (N_X + N_WQKV));
  unsigned short* qbuf = (unsigned short*)(ws + 2 * (N_X + N_WQKV + N_WOUT));
  unsigned short* kbuf = qbuf + N_HBUF;
  unsigned short* vbuf = kbuf + N_HBUF;
  unsigned short* abuf = vbuf + N_HBUF;
  // total workspace: 2*(N_X + N_WQKV + N_WOUT + 4*N_HBUF) = 50,331,648 bytes

  cvt_f32_bf16<<<(int)((N_X    + 255) / 256), 256, 0, stream>>>(x,     xb,  (int)N_X);
  cvt_f32_bf16<<<(int)((N_WQKV + 255) / 256), 256, 0, stream>>>(w_qkv, wqb, (int)N_WQKV);
  cvt_f32_bf16<<<(int)((N_WOUT + 255) / 256), 256, 0, stream>>>(w_out, wob, (int)N_WOUT);

  // QKV:     M=4096 (32 x 128-row blocks), N=3072 (12 x 256-col blocks)
  gemm_qkv_kernel<<<dim3(32, 12), 256, 0, stream>>>(xb, wqb, qbuf, kbuf, vbuf);
  // Attention: 32 (b,h) x 64 query-tiles of 32 rows = 2048 waves / 8 per block
  attn_kernel<<<256, 256, 0, stream>>>(qbuf, kbuf, vbuf, abuf);
  // Out-proj: M=4096, N=1024 (4 x 256-col blocks)
  gemm_out_kernel<<<dim3(32, 4), 256, 0, stream>>>(abuf, wob, out);
}
